// random_forest_net_39694087750207
// MI455X (gfx1250) — matrix operand order repaired and checked by reading, not compiled, not run on hardware
//
#include <hip/hip_runtime.h>

// Soft decision forest, structure-exploiting implementation for gfx1250.
//
// Stage 1 (z): W1 is one-hot per internal node -> recover feat[t,d] once per
//   launch (kernel 1), then z = 10*(x[b,feat] + b1) is a gather, not a GEMM.
// Stage 2 (leaf log-probs): path masks Bpos/Bneg are the deterministic
//   complete-heap paths from the reference's _build_path_masks (RNG-free),
//   8 nonzeros per leaf -> 8-term sum computed analytically.
//   logsig(-z) = logsig(z) - z saves half the transcendentals.
// Stage 3 (leaf values -> classes): genuinely dense [B x T*K]x[T*K x C] GEMM
//   done with v_wmma_f32_16x16x32_f16 (f16 in, f32 accumulate).
//
// Scheduling: Cw fragment loads hoisted to the top of each tree iteration so
// their global latency hides under ~48 transcendentals/thread of stage-1/2.
// Only a wave-uniform Cw prefetch is kept (divergent prefetches generated
// exec-mask bookkeeping that outweighed their value).

typedef __attribute__((ext_vector_type(16))) _Float16 v16h;
typedef __attribute__((ext_vector_type(8)))  _Float16 v8h;
typedef __attribute__((ext_vector_type(8)))  float    v8f;

#define T_TREES  64
#define TDEPTH   8
#define D_NODES  255
#define K_LEAVES 256
#define M_FEAT   512
#define C_CLS    8
#define B_BATCH  4096
#define TILE_B   16

// Kernel 1: recover the one-hot feature index of each internal node from W1.
__global__ void extract_feat_kernel(const float* __restrict__ W1,
                                    int* __restrict__ feat, int n) {
  int i = blockIdx.x * blockDim.x + threadIdx.x;
  if (i < n) {
    if (W1[i] > 0.5f) feat[i >> 9] = (i & (M_FEAT - 1));  // M_FEAT == 512
  }
}

// Kernel 2: fused forest evaluation. One block = 16 batch rows, all 64 trees.
__global__ __launch_bounds__(256) void forest_kernel(
    const float* __restrict__ x, const float* __restrict__ b1,
    const float* __restrict__ Cw, const int* __restrict__ feat,
    float* __restrict__ out) {
  // LDS layout (56 KB static):
  //   xs   [16][512] f32        bytes     0 .. 32767
  //   lsig [16][256][2] f16     bytes 32768 .. 49151  (aliased by red below)
  //   red  [8][16][16] f32      bytes 32768 .. 40959  (after tree loop only)
  //   xc   [16][256] f16        bytes 49152 .. 57343
  __shared__ __align__(16) unsigned char smem[57344];
  float*    xs   = (float*)smem;
  _Float16* lsig = (_Float16*)(smem + 32768);
  float*    red  = (float*)(smem + 32768);
  _Float16* xc   = (_Float16*)(smem + 49152);

  const int tid  = threadIdx.x;
  const int lane = tid & 31;
  const int w    = tid >> 5;            // wave id 0..7 -> K-chunk of 32 leaves
  const int b0   = blockIdx.x * TILE_B;

  // Load 16 x-rows into LDS (coalesced).
  for (int i = tid; i < TILE_B * M_FEAT; i += 256) {
    int r = i >> 9, m = i & (M_FEAT - 1);
    xs[i] = x[(b0 + r) * M_FEAT + m];
  }

  // Per-thread leaf path (leaf k == tid), identical for all trees:
  // bit==0 -> Bpos -> log(1-x) (slot 0), bit==1 -> Bneg -> log(x) (slot 1).
  int idx[TDEPTH];
  {
    int node = 0;
#pragma unroll
    for (int l = 0; l < TDEPTH; ++l) {
      int bit = (tid >> (TDEPTH - 1 - l)) & 1;
      idx[l] = node * 2 + bit;
      node = 2 * node + 1 + bit;
    }
  }

  // WMMA fragment coordinates (wave32 layouts per CDNA5 ISA 7.12.2).
  const int   mrow  = (lane < 16) ? lane : (lane - 16);   // A row / B col
  const int   akb   = w * 32 + ((lane < 16) ? 0 : 8);     // A: K base (elems 0-7)
  const int   bk0   = w * 32 + ((lane < 16) ? 0 : 8);     // B: K base (elems 0-7)
  const int   bcol  = mrow & 7;                           // clamped (always valid)
  const float bmask = (mrow < C_CLS) ? 1.0f : 0.0f;       // zero cols 8..15
  v8f acc = {};                                           // 16x16 f32 C tile

  __syncthreads();

  for (int t = 0; t < T_TREES; ++t) {
    // ---- B fragment: issue Cw global loads first; latency hides under the
    //      stage-1/2 transcendentals below (no LDS dependence).
    //      Unconditional loads from a clamped column, masked arithmetically
    //      (no divergence, no address cndmasks). ----
    v16h bfrag;
    {
      const float* cwt = Cw + t * (K_LEAVES * C_CLS);
#pragma unroll
      for (int e = 0; e < 16; ++e) {
        bfrag[e] = (_Float16)(cwt[(bk0 + (e & 7) + ((e >> 3) << 4)) * C_CLS + bcol] * bmask);
      }
    }
    // Wave-uniform prefetch of next tree's Cw (8 KB, one 256B span per lane).
    if ((w == 0) & (t + 1 < T_TREES)) {
      __builtin_prefetch(Cw + (t + 1) * (K_LEAVES * C_CLS) + lane * 64, 0, 3);
    }

    // ---- stage 1: z + log-sigmoid for node d == tid, 16 rows ----
    if (tid < D_NODES) {
      const int   f  = feat[t * D_NODES + tid];
      const float bb = b1[t * D_NODES + tid];
#pragma unroll 4
      for (int r = 0; r < TILE_B; ++r) {
        float z  = 10.0f * (xs[r * M_FEAT + f] + bb);
        float lg = __logf(1.0f + __expf(-fabsf(z)));
        float lp = fminf(z, 0.0f) - lg;   // log sigmoid(z)   (= log x)
        float ln = lp - z;                // log sigmoid(-z)  (= log(1-x))
        lsig[r * 512 + tid * 2 + 0] = (_Float16)ln;
        lsig[r * 512 + tid * 2 + 1] = (_Float16)lp;
      }
    }
    __syncthreads();

    // ---- stage 2: leaf probability for leaf k == tid, 16 rows ----
#pragma unroll 4
    for (int r = 0; r < TILE_B; ++r) {
      float s = 0.0f;
#pragma unroll
      for (int l = 0; l < TDEPTH; ++l) s += (float)lsig[r * 512 + idx[l]];
      xc[r * K_LEAVES + tid] = (_Float16)__expf(s);
    }
    __syncthreads();

    // ---- stage 3: acc += xc[16 x (w-th 32-chunk)] * Cw[t][chunk][:] ----
    v16h a;
    {
      v8h lo = *(const v8h*)&xc[mrow * K_LEAVES + akb];
      v8h hi = *(const v8h*)&xc[mrow * K_LEAVES + akb + 16];
#pragma unroll
      for (int e = 0; e < 8; ++e) { a[e] = lo[e]; a[e + 8] = hi[e]; }
    }
    acc = __builtin_amdgcn_wmma_f32_16x16x32_f16(false, a, false, bfrag,
                                                 (short)0, acc, false, false);
  }

  __syncthreads();
  // Cross-wave reduction of the 8 partial 16x16 tiles (red aliases lsig).
#pragma unroll
  for (int r = 0; r < 8; ++r) {
    int M = r + ((lane < 16) ? 0 : 8);
    int N = lane & 15;
    red[w * 256 + M * 16 + N] = acc[r];
  }
  __syncthreads();
  if (tid < TILE_B * C_CLS) {
    int m = tid >> 3, n = tid & 7;
    float s = 0.0f;
#pragma unroll
    for (int ww = 0; ww < 8; ++ww) s += red[ww * 256 + m * 16 + n];
    out[(b0 + m) * C_CLS + n] = s * (1.0f / (float)T_TREES);
  }
}

extern "C" void kernel_launch(void* const* d_in, const int* in_sizes, int n_in,
                              void* d_out, int out_size, void* d_ws, size_t ws_size,
                              hipStream_t stream) {
  // setup_inputs order: input, W1, b1, Bpos, Bneg, Cw
  const float* x  = (const float*)d_in[0];
  const float* W1 = (const float*)d_in[1];
  const float* b1 = (const float*)d_in[2];
  const float* Cw = (const float*)d_in[5];
  float* out = (float*)d_out;
  int*   feat = (int*)d_ws;   // T*D ints = 65,280 bytes of scratch

  const int n_w1 = T_TREES * D_NODES * M_FEAT;   // 8,355,840
  extract_feat_kernel<<<(n_w1 + 255) / 256, 256, 0, stream>>>(W1, feat, n_w1);
  forest_kernel<<<B_BATCH / TILE_B, 256, 0, stream>>>(x, b1, Cw, feat, out);
}
